// PointEmbed_38474317037977
// MI455X (gfx1250) — matrix operand order repaired and checked by reading, not compiled, not run on hardware
//
#include <hip/hip_runtime.h>
#include <math.h>

// ---------- types ----------
typedef __attribute__((ext_vector_type(16))) __bf16 v16bf;
typedef __attribute__((ext_vector_type(8)))  float  v8f;
typedef __attribute__((ext_vector_type(4)))  unsigned int u32x4;
typedef __attribute__((ext_vector_type(8)))  int    i32x8;
typedef __attribute__((ext_vector_type(4)))  int    i32x4;

#define BB  4
#define CC  3
#define NN  4096
#define EE  256
#define KNB 15          // neighbors kept (K=16, drop top-1)
#define TCOUNT (BB*NN*KNB)

// ---------- workspace layout (bytes) ----------
#define XN_OFF   0            // xn (B,C,N) f32           196608
#define X2_OFF   196608       // x2 (B,N)   f32           65536
#define BN1_OFF  262144       // scale[3], shift[3]       64
#define SUM_OFF  262208       // 9 atomics sums           64
#define A2D_OFF  262400       // float4[256]              4096
#define IDX_OFF  266496       // idx (B,N,15) i32         983040
#define PE_OFF   1249536      // point_emb (B,E,N) f32    16777216
#define NE_OFF   18026752     // neigh_emb (B,E,N) f32    16777216
#define W2B_OFF  34803968     // swizzled W2b bf16        131072
#define WF_OFF   34935040     // swizzled Wf bf16         262144

// ---------- TDM: 1D async load of `ndwords` dwords into LDS ----------
__device__ __forceinline__ void tdm_load_1d(unsigned long long gaddr,
                                            unsigned int lds_off,
                                            unsigned int ndwords) {
  u32x4 g0;
  g0[0] = 1u;                                           // count=1, user descriptor
  g0[1] = lds_off;                                      // lds_addr
  g0[2] = (unsigned int)(gaddr & 0xFFFFFFFFull);        // global_addr[31:0]
  g0[3] = (unsigned int)((gaddr >> 32) & 0x1FFFFFFu)    // global_addr[56:32]
          | (2u << 30);                                 // type=2 (image)
  i32x8 g1;
  g1[0] = (int)(2u << 16);                              // data_size=2 (4B), mask=0
  g1[1] = (int)((ndwords & 0xFFFFu) << 16);             // tensor_dim0[15:0]
  g1[2] = (int)(((ndwords >> 16) & 0xFFFFu) | (1u << 16)); // dim0 hi, tensor_dim1=1
  g1[3] = (int)((ndwords & 0xFFFFu) << 16);             // tile_dim0 = ndwords
  g1[4] = 1;                                            // tile_dim1=1, tile_dim2=0
  g1[5] = (int)ndwords;                                 // tensor_dim0_stride
  g1[6] = 0;
  g1[7] = 0;
  i32x4 z4 = {0, 0, 0, 0};
  i32x8 z8 = {0, 0, 0, 0, 0, 0, 0, 0};
  // clang-23 / therock-10.0 6-arg form:
  // (g0[4], g1[8], g2[4], g3[4], extra[8], cpol)
  __builtin_amdgcn_tensor_load_to_lds(g0, g1, z4, z4, z8, 0);
}

__device__ __forceinline__ float gelu_exact(float x) {
  return 0.5f * x * (1.0f + erff(x * 0.70710678118654752f));
}

// ============================================================
// K0: zero the atomic-sum scratch
// ============================================================
__global__ void k_zero(float* sums) {
  if (threadIdx.x < 16) sums[threadIdx.x] = 0.0f;
}

// ============================================================
// K1: BN1 stats per channel -> scale/shift
// ============================================================
__global__ __launch_bounds__(256) void k_bn1(const float* __restrict__ x,
                                             const float* __restrict__ g1,
                                             const float* __restrict__ b1,
                                             float* __restrict__ bn1) {
  int c = blockIdx.x, tid = threadIdx.x;
  float s = 0.f, q = 0.f;
  for (int j = tid; j < BB * NN; j += 256) {
    int b = j >> 12, n = j & (NN - 1);
    float v = x[(b * CC + c) * NN + n];
    s += v; q += v * v;
  }
  __shared__ float rs[256], rq[256];
  rs[tid] = s; rq[tid] = q; __syncthreads();
  for (int st = 128; st > 0; st >>= 1) {
    if (tid < st) { rs[tid] += rs[tid + st]; rq[tid] += rq[tid + st]; }
    __syncthreads();
  }
  if (tid == 0) {
    float mu  = rs[0] * (1.0f / (BB * NN));
    float var = rq[0] * (1.0f / (BB * NN)) - mu * mu;
    float sc  = g1[c] * rsqrtf(var + 1e-5f);
    bn1[c]     = sc;
    bn1[3 + c] = b1[c] - mu * sc;
  }
}

// ============================================================
// K2: normalize x -> xn, x2 = sum_c xn^2
// ============================================================
__global__ __launch_bounds__(256) void k_norm(const float* __restrict__ x,
                                              const float* __restrict__ bn1,
                                              float* __restrict__ xn,
                                              float* __restrict__ x2) {
  int j = blockIdx.x * 256 + threadIdx.x;          // 16384
  int b = j >> 12, n = j & (NN - 1);
  float s = 0.f;
#pragma unroll
  for (int c = 0; c < CC; ++c) {
    float v = x[(b * CC + c) * NN + n] * bn1[c] + bn1[3 + c];
    xn[(b * CC + c) * NN + n] = v;
    s += v * v;
  }
  x2[j] = s;
}

// ============================================================
// K3: point embedding  pe[b,e,n] = W1[e,:] . xn[b,:,n] + bw1[e]
// ============================================================
__global__ __launch_bounds__(256) void k_pe(const float* __restrict__ xn,
                                            const float* __restrict__ W1,
                                            const float* __restrict__ bw1,
                                            float* __restrict__ pe) {
  int t = blockIdx.x * 256 + threadIdx.x;          // 4*256*4096
  int n = t & (NN - 1);
  int e = (t >> 12) & (EE - 1);
  int b = t >> 20;
  float r = bw1[e];
#pragma unroll
  for (int c = 0; c < CC; ++c)
    r += W1[e * CC + c] * xn[(b * CC + c) * NN + n];
  pe[t] = r;
}

// ============================================================
// K4: fused distance + top-16 (largest d2, matching jax top_k),
//     drop #1, store 15 neighbor indices. xn rows staged in LDS.
// ============================================================
__global__ __launch_bounds__(256) void k_knn(const float* __restrict__ xn,
                                             const float* __restrict__ x2,
                                             int* __restrict__ idx) {
  __shared__ float sX0[NN], sX1[NN], sX2[NN], sD[NN];   // 64KB
  int blk = blockIdx.x;                // 64 blocks: 16 per batch
  int b = blk >> 4;
  int n = ((blk & 15) << 8) + threadIdx.x;
  for (int j = threadIdx.x; j < NN; j += 256) {
    sX0[j] = xn[(b * CC + 0) * NN + j];
    sX1[j] = xn[(b * CC + 1) * NN + j];
    sX2[j] = xn[(b * CC + 2) * NN + j];
    sD[j]  = x2[b * NN + j];
  }
  __syncthreads();
  float p0 = sX0[n], p1 = sX1[n], p2 = sX2[n], o2 = sD[n];
  float v[16]; int ixv[16];
#pragma unroll
  for (int j = 0; j < 16; ++j) { v[j] = -INFINITY; ixv[j] = 0; }
  for (int m = 0; m < NN; ++m) {
    float d = o2 + sD[m] - 2.0f * (p0 * sX0[m] + p1 * sX1[m] + p2 * sX2[m]);
    if (d > v[15]) {
#pragma unroll
      for (int j = 15; j >= 0; --j) {
        if (j > 0 && v[j - 1] < d) { v[j] = v[j - 1]; ixv[j] = ixv[j - 1]; }
        else if (v[j] < d)         { v[j] = d;        ixv[j] = m; }
      }
    }
  }
  int base = (b * NN + n) * KNB;
#pragma unroll
  for (int k = 1; k < 16; ++k) idx[base + (k - 1)] = ixv[k];
}

// ============================================================
// K5: edge-feature channel sums + second moments (3 + 6 values)
// ============================================================
__global__ __launch_bounds__(256) void k_nstat(const float* __restrict__ xn,
                                               const int* __restrict__ idx,
                                               float* __restrict__ sums) {
  int j = blockIdx.x * 256 + threadIdx.x;          // 16384
  int b = j >> 12, n = j & (NN - 1);
  float a0 = xn[(b * CC + 0) * NN + n];
  float a1 = xn[(b * CC + 1) * NN + n];
  float a2 = xn[(b * CC + 2) * NN + n];
  float s0 = 0, s1 = 0, s2 = 0, q00 = 0, q01 = 0, q02 = 0, q11 = 0, q12 = 0, q22 = 0;
  for (int k = 0; k < KNB; ++k) {
    int m = idx[j * KNB + k];
    float d0 = xn[(b * CC + 0) * NN + m] - a0;
    float d1 = xn[(b * CC + 1) * NN + m] - a1;
    float d2 = xn[(b * CC + 2) * NN + m] - a2;
    s0 += d0; s1 += d1; s2 += d2;
    q00 += d0 * d0; q01 += d0 * d1; q02 += d0 * d2;
    q11 += d1 * d1; q12 += d1 * d2; q22 += d2 * d2;
  }
#pragma unroll
  for (int off = 1; off < 32; off <<= 1) {
    s0 += __shfl_xor(s0, off, 32); s1 += __shfl_xor(s1, off, 32);
    s2 += __shfl_xor(s2, off, 32); q00 += __shfl_xor(q00, off, 32);
    q01 += __shfl_xor(q01, off, 32); q02 += __shfl_xor(q02, off, 32);
    q11 += __shfl_xor(q11, off, 32); q12 += __shfl_xor(q12, off, 32);
    q22 += __shfl_xor(q22, off, 32);
  }
  if ((threadIdx.x & 31) == 0) {
    atomicAdd(&sums[0], s0);  atomicAdd(&sums[1], s1);  atomicAdd(&sums[2], s2);
    atomicAdd(&sums[3], q00); atomicAdd(&sums[4], q01); atomicAdd(&sums[5], q02);
    atomicAdd(&sums[6], q11); atomicAdd(&sums[7], q12); atomicAdd(&sums[8], q22);
  }
}

// ============================================================
// K6: fold BN2a -> W2a -> BN2b into per-e affine (A2[3], d2):
//     gelu input = A2 . neigh + d2
// ============================================================
__global__ __launch_bounds__(256) void k_fold(const float* __restrict__ sums,
                                              const float* __restrict__ W2a,
                                              const float* __restrict__ g2a,
                                              const float* __restrict__ b2a,
                                              const float* __restrict__ g2b,
                                              const float* __restrict__ b2b,
                                              float4* __restrict__ a2d) {
  int e = threadIdx.x;
  const float invT = 1.0f / (float)TCOUNT;
  float m[3]; float C[3][3];
  m[0] = sums[0] * invT; m[1] = sums[1] * invT; m[2] = sums[2] * invT;
  C[0][0] = sums[3] * invT - m[0] * m[0];
  C[0][1] = C[1][0] = sums[4] * invT - m[0] * m[1];
  C[0][2] = C[2][0] = sums[5] * invT - m[0] * m[2];
  C[1][1] = sums[6] * invT - m[1] * m[1];
  C[1][2] = C[2][1] = sums[7] * invT - m[1] * m[2];
  C[2][2] = sums[8] * invT - m[2] * m[2];
  float A[3]; float dd = 0.f;
#pragma unroll
  for (int c = 0; c < 3; ++c) {
    float sc = g2a[c] * rsqrtf(C[c][c] + 1e-5f);
    A[c] = W2a[e * 3 + c] * sc;
    dd  += W2a[e * 3 + c] * (b2a[c] - m[c] * sc);
  }
  float mu2 = A[0] * m[0] + A[1] * m[1] + A[2] * m[2] + dd;
  float var2 = 0.f;
#pragma unroll
  for (int a = 0; a < 3; ++a)
#pragma unroll
    for (int c = 0; c < 3; ++c) var2 += A[a] * A[c] * C[a][c];
  float g = g2b[e] * rsqrtf(var2 + 1e-5f);
  a2d[e] = make_float4(A[0] * g, A[1] * g, A[2] * g, (dd - mu2) * g + b2b[e]);
}

// ============================================================
// K7: swizzle a row-major (M,K) f32 matrix into the bf16
//     WMMA A-fragment order used here (16x32 tiles, k = 16*(lane/16) + i)
// ============================================================
__global__ __launch_bounds__(256) void k_swz(const float* __restrict__ src,
                                             __bf16* __restrict__ dst,
                                             int M, int Kd) {
  int o = blockIdx.x * 256 + threadIdx.x;      // M*Kd total
  int i = o & 15;
  int L = (o >> 4) & 31;
  int tile = o >> 9;
  int MT = M >> 4;
  int mt = tile % MT;
  int kt = tile / MT;
  int mm = mt * 16 + (L & 15);
  int kk = kt * 32 + ((L >> 4) << 4) + i;
  dst[o] = (__bf16)src[mm * Kd + kk];
}

// ============================================================
// K8: fused neighbor branch:  neigh_emb[b,f,n] =
//       max_k  W2b @ gelu(A2 . (xn[:,idx]-xn) + d2)
//     - 256x256 GEMM over 16 cols/point (col0 duplicates k=0)
//     - W2b streamed LDS via TDM double-buffered 16KB slabs
//     - bf16 WMMA, f32 accum
// ============================================================
__global__ __launch_bounds__(256) void k_gemm_knn(
    const float* __restrict__ xn, const int* __restrict__ idx,
    const float4* __restrict__ a2d, const __bf16* __restrict__ w2b_sw,
    float* __restrict__ ne) {
  __shared__ __align__(64) __bf16 sW[2][8192];   // 2 x 16KB slabs
  __shared__ float sN0[128], sN1[128], sN2[128];
  __shared__ float4 sA2[256];
  int tid = threadIdx.x, lane = tid & 31, wave = tid >> 5;
  int blk = blockIdx.x;                 // 2048 = B * N/8
  int b = blk >> 9;
  int n0 = (blk & 511) << 3;

  sA2[tid] = a2d[tid];
  if (tid < 128) {
    int p = tid >> 4, t = tid & 15;
    int n = n0 + p;
    int kk = (t == 0) ? 0 : (t - 1);          // duplicate neighbor 0: max-safe pad
    int m = idx[(b * NN + n) * KNB + kk];
    sN0[tid] = xn[(b * CC + 0) * NN + m] - xn[(b * CC + 0) * NN + n];
    sN1[tid] = xn[(b * CC + 1) * NN + m] - xn[(b * CC + 1) * NN + n];
    sN2[tid] = xn[(b * CC + 2) * NN + m] - xn[(b * CC + 2) * NN + n];
  }
  if (wave == 0) {
    tdm_load_1d((unsigned long long)(size_t)w2b_sw,
                (unsigned int)(size_t)&sW[0][0], 4096);
  }
  __syncthreads();

  int myCol = wave * 16 + (lane & 15);
  float c0 = sN0[myCol], c1 = sN1[myCol], c2 = sN2[myCol];

  v8f zero = {0.f, 0.f, 0.f, 0.f, 0.f, 0.f, 0.f, 0.f};
  v8f acc[16];
#pragma unroll
  for (int mt = 0; mt < 16; ++mt) acc[mt] = zero;

  for (int kt = 0; kt < 8; ++kt) {
    if (wave == 0) {
      if (kt < 7) {
        tdm_load_1d((unsigned long long)(size_t)w2b_sw + (kt + 1) * 16384u,
                    (unsigned int)(size_t)&sW[(kt + 1) & 1][0], 4096);
        __builtin_amdgcn_s_wait_tensorcnt(1);
      } else {
        __builtin_amdgcn_s_wait_tensorcnt(0);
      }
    }
    __syncthreads();

    // B fragment (32x16, K=e block): gelu of folded affine, in-register
    v16bf bfrag;
    int ebase = kt * 32 + ((lane >> 4) << 4);
#pragma unroll
    for (int i = 0; i < 16; ++i) {
      float4 a = sA2[ebase + i];
      float pre = fmaf(a.x, c0, fmaf(a.y, c1, fmaf(a.z, c2, a.w)));
      bfrag[i] = (__bf16)gelu_exact(pre);
    }
    const __bf16* slab = &sW[kt & 1][0];
#pragma unroll
    for (int mt = 0; mt < 16; ++mt) {
      v16bf afrag = *(const v16bf*)(slab + (mt * 32 + lane) * 16);
      acc[mt] = __builtin_amdgcn_wmma_f32_16x16x32_bf16(
          false, afrag, false, bfrag, (short)0, acc[mt], false, false);
    }
    __syncthreads();
  }

  // max over this wave's 16 columns (all belong to point n0+wave)
  int n = n0 + wave;
#pragma unroll
  for (int mt = 0; mt < 16; ++mt) {
#pragma unroll
    for (int r = 0; r < 8; ++r) {
      float v = acc[mt][r];
      v = fmaxf(v, __shfl_xor(v, 1, 32));
      v = fmaxf(v, __shfl_xor(v, 2, 32));
      v = fmaxf(v, __shfl_xor(v, 4, 32));
      v = fmaxf(v, __shfl_xor(v, 8, 32));
      if ((lane & 15) == 0) {
        int f = mt * 16 + r + ((lane >> 4) << 3);
        ne[((size_t)(b * EE + f)) * NN + n] = v;
      }
    }
  }
}

// ============================================================
// K9: final GEMM  out = Wf @ [point_emb ; neigh_emb] + bf
//     bf16 WMMA, M=256 K=512, 128 cols per block
// ============================================================
__global__ __launch_bounds__(256) void k_fc(
    const float* __restrict__ pe, const float* __restrict__ ne,
    const __bf16* __restrict__ wf_sw, const float* __restrict__ bias,
    float* __restrict__ out) {
  int tid = threadIdx.x, lane = tid & 31, wave = tid >> 5;
  int blk = blockIdx.x;                 // 128 = B * 32
  int b = blk >> 5;
  int n = ((blk & 31) << 7) + wave * 16 + (lane & 15);

  v8f zero = {0.f, 0.f, 0.f, 0.f, 0.f, 0.f, 0.f, 0.f};
  v8f acc[16];
#pragma unroll
  for (int mt = 0; mt < 16; ++mt) acc[mt] = zero;

  for (int kt = 0; kt < 16; ++kt) {
    v16bf bfrag;
    int fbase = kt * 32 + ((lane >> 4) << 4);
#pragma unroll
    for (int i = 0; i < 16; ++i) {
      int f = fbase + i;
      float v = (f < EE) ? pe[((size_t)(b * EE + f)) * NN + n]
                         : ne[((size_t)(b * EE + (f - EE))) * NN + n];
      bfrag[i] = (__bf16)v;
    }
#pragma unroll
    for (int mt = 0; mt < 16; ++mt) {
      v16bf afrag = *(const v16bf*)(wf_sw + ((kt * 16 + mt) * 32 + lane) * 16);
      acc[mt] = __builtin_amdgcn_wmma_f32_16x16x32_bf16(
          false, afrag, false, bfrag, (short)0, acc[mt], false, false);
    }
  }
#pragma unroll
  for (int mt = 0; mt < 16; ++mt) {
#pragma unroll
    for (int r = 0; r < 8; ++r) {
      int o = mt * 16 + r + ((lane >> 4) << 3);
      out[((size_t)(b * EE + o)) * NN + n] = acc[mt][r] + bias[o];
    }
  }
}

// ============================================================
extern "C" void kernel_launch(void* const* d_in, const int* in_sizes, int n_in,
                              void* d_out, int out_size, void* d_ws, size_t ws_size,
                              hipStream_t stream) {
  (void)in_sizes; (void)n_in; (void)out_size; (void)ws_size;
  const float* x   = (const float*)d_in[0];
  const float* g1  = (const float*)d_in[1];
  const float* b1  = (const float*)d_in[2];
  const float* W1  = (const float*)d_in[3];
  const float* bw1 = (const float*)d_in[4];
  const float* g2a = (const float*)d_in[5];
  const float* b2a = (const float*)d_in[6];
  const float* W2a = (const float*)d_in[7];
  const float* g2b = (const float*)d_in[8];
  const float* b2b = (const float*)d_in[9];
  const float* W2b = (const float*)d_in[10];
  const float* Wf  = (const float*)d_in[11];
  const float* bfv = (const float*)d_in[12];

  char* wsb = (char*)d_ws;
  float*  xn     = (float*)(wsb + XN_OFF);
  float*  x2     = (float*)(wsb + X2_OFF);
  float*  bn1    = (float*)(wsb + BN1_OFF);
  float*  sums   = (float*)(wsb + SUM_OFF);
  float4* a2d    = (float4*)(wsb + A2D_OFF);
  int*    idx    = (int*)(wsb + IDX_OFF);
  float*  pe     = (float*)(wsb + PE_OFF);
  float*  ne     = (float*)(wsb + NE_OFF);
  __bf16* w2b_sw = (__bf16*)(wsb + W2B_OFF);
  __bf16* wf_sw  = (__bf16*)(wsb + WF_OFF);
  float*  out    = (float*)d_out;

  k_zero<<<1, 32, 0, stream>>>(sums);
  k_bn1<<<3, 256, 0, stream>>>(x, g1, b1, bn1);
  k_norm<<<64, 256, 0, stream>>>(x, bn1, xn, x2);
  k_pe<<<16384, 256, 0, stream>>>(xn, W1, bw1, pe);
  k_knn<<<64, 256, 0, stream>>>(xn, x2, idx);
  k_nstat<<<64, 256, 0, stream>>>(xn, idx, sums);
  k_fold<<<1, 256, 0, stream>>>(sums, W2a, g2a, b2a, g2b, b2b, a2d);
  k_swz<<<256, 256, 0, stream>>>(W2b, w2b_sw, 256, 256);
  k_swz<<<512, 256, 0, stream>>>(Wf, wf_sw, 256, 512);
  k_gemm_knn<<<2048, 256, 0, stream>>>(xn, idx, a2d, w2b_sw, ne);
  k_fc<<<128, 256, 0, stream>>>(pe, ne, wf_sw, bfv, ne == pe ? (float*)d_out : out);
}
